// Playlist_Model_27900107555446
// MI455X (gfx1250) — matrix operand order repaired and checked by reading, not compiled, not run on hardware
//
#include <hip/hip_runtime.h>

// ---------------------------------------------------------------------------
// Model dims
// ---------------------------------------------------------------------------
#define BATCH 8192
#define DEMB  128
#define LLEN  5
#define FDIM  1920      // 15 * 128
#define PPAD  128       // cross rank 100 padded to 128
#define H0    512
#define H1    256
#define H2    128

#define LDSW  40        // padded LDS row stride in halfwords (80B, bank-conflict free)

typedef __attribute__((ext_vector_type(16))) __bf16 v16bf;
typedef __attribute__((ext_vector_type(8)))  float  v8f;

__device__ __forceinline__ unsigned short f2bf(float f) {
  union { float f; unsigned u; } x; x.f = f;
  unsigned r = x.u + 0x7FFFu + ((x.u >> 16) & 1u);   // round-to-nearest-even
  return (unsigned short)(r >> 16);
}

// ---------------------------------------------------------------------------
// gfx1250 async global->LDS copy (signature: (v4i AS1*, v4i AS3*, imm, imm))
// ---------------------------------------------------------------------------
#if defined(__has_builtin)
#  if __has_builtin(__builtin_amdgcn_global_load_async_to_lds_b128)
#    define HAVE_ASYNC_LDS 1
#  endif
#endif
#ifndef HAVE_ASYNC_LDS
#  define HAVE_ASYNC_LDS 0
#endif

#if HAVE_ASYNC_LDS
typedef __attribute__((vector_size(16))) int v4i_raw;
typedef __attribute__((address_space(1))) v4i_raw* gptr_v4i;
typedef __attribute__((address_space(3))) v4i_raw* lptr_v4i;
__device__ __forceinline__ lptr_v4i to_lds(const void* p) {
  return (lptr_v4i)(unsigned int)(unsigned long long)p;
}
__device__ __forceinline__ gptr_v4i to_glb(const void* p) {
  return (gptr_v4i)(unsigned long long)p;
}
#endif

// ---------------------------------------------------------------------------
// 1) Embedding gather + concat -> x0 (fp32 for cross residual, bf16 for GEMMs)
// ---------------------------------------------------------------------------
struct GatherArgs {
  const int*   idx1[6];
  const int*   idxL[9];
  const float* tab1[6];
  const float* tabL[9];
};

__global__ __launch_bounds__(DEMB) void gather_concat_kernel(
    GatherArgs ga, float* __restrict__ x0f, unsigned short* __restrict__ x0b) {
  const int b = blockIdx.x;
  const int d = threadIdx.x;                 // 0..127
  const size_t ro = (size_t)b * FDIM;

#pragma unroll
  for (int s = 0; s < 6; ++s) {
    const int idx = ga.idx1[s][b];
    const float v = ga.tab1[s][(size_t)idx * DEMB + d];
    x0f[ro + s * DEMB + d] = v;
    x0b[ro + s * DEMB + d] = f2bf(v);
  }

#pragma unroll
  for (int j = 0; j < 9; ++j) {
    const bool masked = (j < 5) || (j == 8);  // artist_pop/followers/track_pop: plain mean
    float sum = 0.f, cnt = 0.f;
#pragma unroll
    for (int l = 0; l < LLEN; ++l) {
      const int idx = ga.idxL[j][b * LLEN + l];
      const float w = (!masked || idx != 0) ? 1.f : 0.f;
      sum += w * ga.tabL[j][(size_t)idx * DEMB + d];
      cnt += w;
    }
    const float v = sum / fmaxf(cnt, 1.f);
    const int s = 6 + j;
    x0f[ro + s * DEMB + d] = v;
    x0b[ro + s * DEMB + d] = f2bf(v);
  }
}

// ---------------------------------------------------------------------------
// 2) fp32 -> bf16 convert, TRANSPOSE to [N,K], zero pad.
//    src is [sK, sN] row-major; dst is [dN, dK] row-major (K contiguous).
// ---------------------------------------------------------------------------
__global__ void cvt_t_pad_kernel(const float* __restrict__ src,
                                 unsigned short* __restrict__ dst,
                                 int sK, int sN, int dK, int dN) {
  const int i = blockIdx.x * blockDim.x + threadIdx.x;
  if (i >= dN * dK) return;
  const int n = i / dK, k = i % dK;
  const float v = (k < sK && n < sN) ? src[(size_t)k * sN + n] : 0.f;
  dst[i] = f2bf(v);
}

// ---------------------------------------------------------------------------
// 3) Tiled bf16 WMMA GEMM: C[M,N] = A[M,K] * Bt[N,K]^T   (+ fused epilogues)
//    Block 256 threads (8 waves), 64x64 output tile, K-step 32,
//    double-buffered async global->LDS staging for both operands.
//    MODE 0: store bf16
//    MODE 1: +bias, ReLU, store bf16
//    MODE 2: +bias, cross combine v = x0*(acc+b)+x0, store bf16 (needs X0 f32)
//    MODE 3: +bias, store fp32
// ---------------------------------------------------------------------------
template <int MODE>
__global__ __launch_bounds__(256) void gemm_wmma_kernel(
    const unsigned short* __restrict__ A, const unsigned short* __restrict__ Bt,
    unsigned short* __restrict__ Obf, float* __restrict__ Of,
    const float* __restrict__ bias, const float* __restrict__ X0,
    int M, int N, int K) {
  __shared__ __align__(16) unsigned short As[2][64 * LDSW];  // [m][k]
  __shared__ __align__(16) unsigned short Bs[2][64 * LDSW];  // [n][k]

  const int tid  = threadIdx.x;
  const int wave = tid >> 5;
  const int lane = tid & 31;
  const int waveM = wave & 3;        // 0..3  -> 16-row slice
  const int waveN = wave >> 2;       // 0..1  -> 32-col slice
  const int bm = blockIdx.y * 64;
  const int bn = blockIdx.x * 64;

  const int mrow = waveM * 16 + (lane & 15);   // A fragment row in LDS tile
  const int ab   = (lane >> 4) << 3;           // A uint sub-offset: 0 or 8
  const int n0   = waveN * 32 + (lane & 15);   // B fragment col (tile 0)
  const int bb   = (lane >> 4) << 3;           // B uint sub-offset: 0 or 8

  const int sr = tid >> 2;                     // staging: row within 64-row tile
  const int sc = (tid & 3) * 8;                // staging: halfword col (16B chunk)

  v8f acc0 = {0.f, 0.f, 0.f, 0.f, 0.f, 0.f, 0.f, 0.f};
  v8f acc1 = {0.f, 0.f, 0.f, 0.f, 0.f, 0.f, 0.f, 0.f};

#if HAVE_ASYNC_LDS
  auto stage = [&](int buf, int k0) {
    __builtin_amdgcn_global_load_async_to_lds_b128(
        to_glb(A + (size_t)(bm + sr) * K + k0 + sc),
        to_lds(&As[buf][sr * LDSW + sc]), 0, 0);
    __builtin_amdgcn_global_load_async_to_lds_b128(
        to_glb(Bt + (size_t)(bn + sr) * K + k0 + sc),
        to_lds(&Bs[buf][sr * LDSW + sc]), 0, 0);
  };
#else
  auto stage = [&](int buf, int k0) {
    const unsigned* ga = (const unsigned*)(A + (size_t)(bm + sr) * K + k0 + sc);
    const unsigned* gb = (const unsigned*)(Bt + (size_t)(bn + sr) * K + k0 + sc);
    unsigned* la = (unsigned*)&As[buf][sr * LDSW + sc];
    unsigned* lb = (unsigned*)&Bs[buf][sr * LDSW + sc];
#pragma unroll
    for (int i = 0; i < 4; ++i) { la[i] = ga[i]; lb[i] = gb[i]; }
  };
#endif

  const int T = K >> 5;      // K/32 tiles (K is always a multiple of 32)
  stage(0, 0);
  for (int t = 0; t < T; ++t) {
    const int cur = t & 1;
    if (t + 1 < T) {
      stage(1 - cur, (t + 1) << 5);
#if HAVE_ASYNC_LDS
      asm volatile("s_wait_asynccnt 2" ::: "memory");  // tile t done (in-order)
#endif
    } else {
#if HAVE_ASYNC_LDS
      asm volatile("s_wait_asynccnt 0" ::: "memory");
#endif
    }
    __syncthreads();

    union { v16bf v; unsigned u[8]; } afrag, bfrag0, bfrag1;
    const unsigned* Arow  = (const unsigned*)(&As[cur][mrow * LDSW]);
    const unsigned* Brow0 = (const unsigned*)(&Bs[cur][n0 * LDSW]);
    const unsigned* Brow1 = (const unsigned*)(&Bs[cur][(n0 + 16) * LDSW]);
#pragma unroll
    for (int i = 0; i < 8; ++i) {
      afrag.u[i]     = Arow[ab + i];          // K = kg*16 + 2i (,+1)
    }
#pragma unroll
    for (int i = 0; i < 8; ++i) {
      bfrag0.u[i] = Brow0[bb + i];            // K = kg*16 + 2i (,+1)
      bfrag1.u[i] = Brow1[bb + i];
    }
    acc0 = __builtin_amdgcn_wmma_f32_16x16x32_bf16(
        false, afrag.v, false, bfrag0.v, (short)0, acc0, false, false);
    acc1 = __builtin_amdgcn_wmma_f32_16x16x32_bf16(
        false, afrag.v, false, bfrag1.v, (short)0, acc1, false, false);
    __syncthreads();   // all reads of buf[cur] done -> safe to prefetch into it
  }

  // ---- epilogue: C/D layout -> m = base + r + 8*(lane>=16), n = lane%16 ----
  const int mBase = bm + waveM * 16 + ((lane >> 4) << 3);
  const int gn0 = bn + waveN * 32 + (lane & 15);
  const int gn1 = gn0 + 16;
#pragma unroll
  for (int r = 0; r < 8; ++r) {
    const int m = mBase + r;
    float v0 = acc0[r];
    float v1 = acc1[r];
    if (MODE == 0) {
      Obf[(size_t)m * N + gn0] = f2bf(v0);
      Obf[(size_t)m * N + gn1] = f2bf(v1);
    } else if (MODE == 1) {
      v0 = fmaxf(v0 + bias[gn0], 0.f);
      v1 = fmaxf(v1 + bias[gn1], 0.f);
      Obf[(size_t)m * N + gn0] = f2bf(v0);
      Obf[(size_t)m * N + gn1] = f2bf(v1);
    } else if (MODE == 2) {
      const float x0a = X0[(size_t)m * N + gn0];
      const float x0c = X0[(size_t)m * N + gn1];
      v0 = x0a * (v0 + bias[gn0]) + x0a;
      v1 = x0c * (v1 + bias[gn1]) + x0c;
      Obf[(size_t)m * N + gn0] = f2bf(v0);
      Obf[(size_t)m * N + gn1] = f2bf(v1);
    } else {  // MODE == 3
      Of[(size_t)m * N + gn0] = v0 + bias[gn0];
      Of[(size_t)m * N + gn1] = v1 + bias[gn1];
    }
  }
}

// ---------------------------------------------------------------------------
// 4) row-wise L2 normalize: out = h * rsqrt(max(sum(h*h), 1e-12))
// ---------------------------------------------------------------------------
__global__ __launch_bounds__(H2) void l2norm_kernel(const float* __restrict__ h,
                                                    float* __restrict__ out) {
  const int b = blockIdx.x, t = threadIdx.x;      // 128 threads = 4 waves
  const float v = h[(size_t)b * H2 + t];
  float s = v * v;
#pragma unroll
  for (int off = 16; off > 0; off >>= 1) s += __shfl_xor(s, off, 32);
  __shared__ float red[4];
  if ((t & 31) == 0) red[t >> 5] = s;
  __syncthreads();
  const float tot = red[0] + red[1] + red[2] + red[3];
  out[(size_t)b * H2 + t] = v * rsqrtf(fmaxf(tot, 1e-12f));
}

// ---------------------------------------------------------------------------
// Launch
// ---------------------------------------------------------------------------
extern "C" void kernel_launch(void* const* d_in, const int* in_sizes, int n_in,
                              void* d_out, int out_size, void* d_ws, size_t ws_size,
                              hipStream_t stream) {
  (void)in_sizes; (void)n_in; (void)out_size; (void)ws_size;

  // ---- inputs (setup_inputs() dict order) ----
  GatherArgs ga;
  for (int i = 0; i < 6; ++i) ga.idx1[i] = (const int*)d_in[i];
  for (int i = 0; i < 9; ++i) ga.idxL[i] = (const int*)d_in[6 + i];
  for (int i = 0; i < 6; ++i) ga.tab1[i] = (const float*)d_in[15 + i];
  for (int i = 0; i < 9; ++i) ga.tabL[i] = (const float*)d_in[21 + i];
  const float* cross_V = (const float*)d_in[30];  // [1920,100]
  const float* cross_U = (const float*)d_in[31];  // [100,1920]
  const float* cross_b = (const float*)d_in[32];  // [1920]
  const float* W0 = (const float*)d_in[33];       // [1920,512]
  const float* b0 = (const float*)d_in[34];
  const float* W1 = (const float*)d_in[35];       // [512,256]
  const float* b1 = (const float*)d_in[36];
  const float* W2 = (const float*)d_in[37];       // [256,128]
  const float* b2 = (const float*)d_in[38];

  // ---- workspace carving ----
  char* p = (char*)d_ws;
  auto carve = [&](size_t bytes) {
    void* r = (void*)p;
    p += (bytes + 255) & ~(size_t)255;
    return r;
  };
  float*          x0f   = (float*)         carve((size_t)BATCH * FDIM * 4);
  unsigned short* x0b   = (unsigned short*)carve((size_t)BATCH * FDIM * 2);
  unsigned short* Vt    = (unsigned short*)carve((size_t)PPAD * FDIM * 2);  // [N=128,K=1920]
  unsigned short* Ut    = (unsigned short*)carve((size_t)FDIM * PPAD * 2);  // [N=1920,K=128]
  unsigned short* W0t   = (unsigned short*)carve((size_t)H0 * FDIM * 2);    // [512,1920]
  unsigned short* W1t   = (unsigned short*)carve((size_t)H1 * H0 * 2);      // [256,512]
  unsigned short* W2t   = (unsigned short*)carve((size_t)H2 * H1 * 2);      // [128,256]
  unsigned short* xv    = (unsigned short*)carve((size_t)BATCH * PPAD * 2);
  unsigned short* crs   = (unsigned short*)carve((size_t)BATCH * FDIM * 2);
  unsigned short* h0    = (unsigned short*)carve((size_t)BATCH * H0 * 2);
  unsigned short* h1    = (unsigned short*)carve((size_t)BATCH * H1 * 2);
  float*          h2    = (float*)         carve((size_t)BATCH * H2 * 4);

  // ---- 1) gather/concat ----
  gather_concat_kernel<<<BATCH, DEMB, 0, stream>>>(ga, x0f, x0b);

  // ---- 2) weight converts: bf16 + transpose to [N,K] (+zero pad cross) ----
  auto cvtT = [&](const float* s, unsigned short* d, int sK, int sN, int dK, int dN) {
    const int tot = dN * dK;
    cvt_t_pad_kernel<<<(tot + 255) / 256, 256, 0, stream>>>(s, d, sK, sN, dK, dN);
  };
  cvtT(cross_V, Vt, FDIM, 100, FDIM, PPAD);
  cvtT(cross_U, Ut, 100, FDIM, PPAD, FDIM);
  cvtT(W0, W0t, FDIM, H0, FDIM, H0);
  cvtT(W1, W1t, H0, H1, H0, H1);
  cvtT(W2, W2t, H1, H2, H1, H2);

  // ---- 3) GEMM chain ----
  const dim3 blk(256);
  // xv = x0 @ V                [8192,128]
  gemm_wmma_kernel<0><<<dim3(PPAD / 64, BATCH / 64), blk, 0, stream>>>(
      x0b, Vt, xv, nullptr, nullptr, nullptr, BATCH, PPAD, FDIM);
  // cross = x0 * (xv @ U + b) + x0       [8192,1920]
  gemm_wmma_kernel<2><<<dim3(FDIM / 64, BATCH / 64), blk, 0, stream>>>(
      xv, Ut, crs, nullptr, cross_b, x0f, BATCH, FDIM, PPAD);
  // h0 = relu(cross @ W0 + b0)            [8192,512]
  gemm_wmma_kernel<1><<<dim3(H0 / 64, BATCH / 64), blk, 0, stream>>>(
      crs, W0t, h0, nullptr, b0, nullptr, BATCH, H0, FDIM);
  // h1 = relu(h0 @ W1 + b1)               [8192,256]
  gemm_wmma_kernel<1><<<dim3(H1 / 64, BATCH / 64), blk, 0, stream>>>(
      h0, W1t, h1, nullptr, b1, nullptr, BATCH, H1, H0);
  // h2 = h1 @ W2 + b2 (fp32)              [8192,128]
  gemm_wmma_kernel<3><<<dim3(H2 / 64, BATCH / 64), blk, 0, stream>>>(
      h1, W2t, nullptr, h2, b2, nullptr, BATCH, H2, H1);

  // ---- 4) L2 normalize -> d_out ----
  l2norm_kernel<<<BATCH, H2, 0, stream>>>(h2, (float*)d_out);
}
